// SCGA_85899346149
// MI455X (gfx1250) — matrix operand order repaired and checked by reading, not compiled, not run on hardware
//
#include <hip/hip_runtime.h>
#include <math.h>

// ---------------------------------------------------------------------------
// SCGA block for MI455X (gfx1250, wave32).
// Heavy op = 3x3x3 conv (58 GFLOP) -> implicit GEMM on v_wmma_f32_16x16x32_f16.
// Everything else is bandwidth-bound; gx (128MB) is L2-resident (192MB L2).
// Simplifications proven from the reference:
//   * softmax over a size-1 axis == 1  -> weights = sigmoid(sum_c(x1+x2))
//   * b3 cancels inside instance_norm -> dropped
//   * att factorizes: sigH[d]*sigW[h]*sigD[w] -> x1 never materialized
// ---------------------------------------------------------------------------

#define SZ   64
#define CG   32
#define NG   4                    // bg = b*GROUP
#define NSP  (SZ*SZ*SZ)           // 262144 spatial
#define EPSV 1e-5f

typedef _Float16 v8h  __attribute__((ext_vector_type(8)));
typedef _Float16 v16h __attribute__((ext_vector_type(16)));
typedef float    v8f  __attribute__((ext_vector_type(8)));

union H16 { v16h v; v8h h[2]; };

// workspace layout (float offsets)
#define OFF_Z    0                                   // 33554432 f  (conv out)
#define OFF_POOL 33554432                            // 3*4*32*64 = 24576 f
#define OFF_SIG  (OFF_POOL + 24576)                  // 24576 f
#define OFF_M1   (OFF_SIG + 24576)                   // 128 f
#define OFF_I1   (OFF_M1 + 128)
#define OFF_M2   (OFF_I1 + 128)
#define OFF_I2   (OFF_M2 + 128)
#define OFF_WPK  (OFF_I2 + 128)                      // 27648 f16 (A fragments)

// --- K0: repack w3 (f32 [co][ci][kd][kh][kw]) into per-lane WMMA A-fragment
//     order (ISA 7.12.2 16-bit A 16x32): lane<16 K={0..7,16..23}, lane>=16
//     K={8..15,24..31}. Layout: wp[t][mtile][lane][16] -> 2x b128 per fragment.
__global__ void k0_repack_w3(const float* __restrict__ w3, _Float16* __restrict__ wp) {
    int idx = blockIdx.x * 256 + threadIdx.x;
    if (idx >= 27 * 2 * 32 * 16) return;
    int kk   = idx & 15;
    int lane = (idx >> 4) & 31;
    int mt   = (idx >> 9) & 1;
    int t    = idx >> 10;
    int ci   = (lane < 16) ? ((kk < 8) ? kk : kk + 8)
                           : ((kk < 8) ? kk + 8 : kk + 16);
    int co   = mt * 16 + (lane & 15);
    wp[idx] = (_Float16)w3[(co * CG + ci) * 27 + t];
}

// --- K1: directional pooling 0.5*mean + 0.5*max over the two non-kept axes.
__global__ __launch_bounds__(256) void k1_pool(const float* __restrict__ gx,
                                               float* __restrict__ pool) {
    int s = blockIdx.x, a = blockIdx.y, gc = blockIdx.z;
    const float* base = gx + (size_t)gc * NSP;
    int tid = threadIdx.x;
    float sum = 0.f, mx = -INFINITY;
    for (int i = tid; i < SZ * SZ; i += 256) {
        int p = i >> 6, q = i & 63;
        int n;
        if      (a == 0) n = (s << 12) + (p << 6) + q;   // keep d
        else if (a == 1) n = (p << 12) + (s << 6) + q;   // keep h
        else             n = (p << 12) + (q << 6) + s;   // keep w
        float v = base[n];
        sum += v; mx = fmaxf(mx, v);
    }
    __shared__ float rs[256], rm[256];
    rs[tid] = sum; rm[tid] = mx; __syncthreads();
    for (int off = 128; off > 0; off >>= 1) {
        if (tid < off) { rs[tid] += rs[tid+off]; rm[tid] = fmaxf(rm[tid], rm[tid+off]); }
        __syncthreads();
    }
    if (tid == 0)
        pool[a * (NG*CG*SZ) + gc * SZ + s] =
            0.5f * (rs[0] / (float)(SZ*SZ)) + 0.5f * rm[0];
}

// --- K2: 1x1 conv (32x32 x 192) + instance-norm(192) + leaky-relu + sigmoid.
__global__ __launch_bounds__(256) void k2_mlp(const float* __restrict__ w1,
                                              const float* __restrict__ b1,
                                              const float* __restrict__ pool,
                                              float* __restrict__ sig) {
    int g = blockIdx.x, tid = threadIdx.x;
    __shared__ float cat[CG * 192];
    __shared__ float wl[CG * CG];
    __shared__ float hw[CG * 192];
    __shared__ float mL[CG], iL[CG];
    for (int i = tid; i < CG * CG; i += 256) wl[i] = w1[i];
    for (int i = tid; i < CG * 192; i += 256) {
        int c = i / 192, s = i % 192;
        cat[i] = pool[(s >> 6) * (NG*CG*SZ) + (g*CG + c) * SZ + (s & 63)];
    }
    __syncthreads();
    for (int i = tid; i < CG * 192; i += 256) {
        int o = i / 192, s = i % 192;
        float acc = b1[o];
        #pragma unroll
        for (int c = 0; c < CG; ++c) acc += wl[o*CG + c] * cat[c*192 + s];
        hw[i] = acc;
    }
    __syncthreads();
    if (tid < CG) {
        float s1 = 0.f, s2 = 0.f;
        for (int s = 0; s < 192; ++s) { float v = hw[tid*192 + s]; s1 += v; s2 += v*v; }
        float m = s1 / 192.f;
        mL[tid] = m; iL[tid] = rsqrtf(s2 / 192.f - m*m + EPSV);
    }
    __syncthreads();
    for (int i = tid; i < CG * 192; i += 256) {
        int o = i / 192, s = i % 192;
        float v = (hw[i] - mL[o]) * iL[o];
        v = (v >= 0.f) ? v : 0.01f * v;                  // leaky_relu
        sig[(s >> 6) * (NG*CG*SZ) + (g*CG + o) * SZ + (s & 63)] =
            1.f / (1.f + __expf(-v));                    // sigmoid
    }
}

// --- K3: per-(g,c) mean / rsqrt-var of gx*att (att from factored sigmoids).
__global__ __launch_bounds__(256) void k3_stats_att(const float* __restrict__ gx,
                                                    const float* __restrict__ sig,
                                                    float* __restrict__ m1,
                                                    float* __restrict__ i1) {
    int gc = blockIdx.x, tid = threadIdx.x;
    __shared__ float sH[SZ], sW[SZ], sD[SZ];
    if (tid < 64)       sH[tid]       = sig[0*(NG*CG*SZ) + gc*SZ + tid];
    else if (tid < 128) sW[tid - 64]  = sig[1*(NG*CG*SZ) + gc*SZ + tid - 64];
    else if (tid < 192) sD[tid - 128] = sig[2*(NG*CG*SZ) + gc*SZ + tid - 128];
    __syncthreads();
    const float* base = gx + (size_t)gc * NSP;
    float s1 = 0.f, s2 = 0.f;
    for (int n = tid; n < NSP; n += 256) {
        int d = n >> 12, h = (n >> 6) & 63, w = n & 63;
        float v = base[n] * sH[d] * sW[h] * sD[w];
        s1 += v; s2 += v * v;
    }
    __shared__ float r1[256], r2[256];
    r1[tid] = s1; r2[tid] = s2; __syncthreads();
    for (int off = 128; off > 0; off >>= 1) {
        if (tid < off) { r1[tid] += r1[tid+off]; r2[tid] += r2[tid+off]; }
        __syncthreads();
    }
    if (tid == 0) {
        float m = r1[0] / (float)NSP;
        m1[gc] = m; i1[gc] = rsqrtf(r2[0] / (float)NSP - m*m + EPSV);
    }
}

// --- K4: 3x3x3 conv as implicit GEMM on v_wmma_f32_16x16x32_f16.
//     Block = one (g,d,h) row. 8 waves = 2 co-tiles x 4 w-tiles.
//     K loop = 27 taps, each tap contracts all 32 input channels in one WMMA.
//     LDS tile is channel-LAST so each lane's 16-ci B slice is two 16-byte runs
//     32 B apart (2x ds_load_b128); A fragments stream from the L2-hot repack.
__global__ __launch_bounds__(256) void k4_conv_wmma(const float* __restrict__ gx,
                                                    const _Float16* __restrict__ wp,
                                                    float* __restrict__ z) {
    int bid = blockIdx.x;
    int h = bid & 63, d = (bid >> 6) & 63, g = bid >> 12;
    __shared__ _Float16 tile[9 * 66 * CG];               // 38016 B
    int tid = threadIdx.x;
    const float* gbase = gx + (size_t)g * CG * NSP;
    for (int idx = tid; idx < 9 * 66 * CG; idx += 256) {
        int ci = idx & 31;
        int rest = idx >> 5;
        int wi = rest % 66 - 1;                          // -1..64 (halo)
        int kk = rest / 66;                              // kd*3+kh
        int din = d + kk / 3 - 1, hin = h + kk % 3 - 1;
        float v = 0.f;
        if (((unsigned)din < SZ) & ((unsigned)hin < SZ) & ((unsigned)wi < SZ))
            v = gbase[((size_t)ci * SZ + din) * (SZ*SZ) + hin * SZ + wi];
        tile[idx] = (_Float16)v;
    }
    __syncthreads();

    int lane = tid & 31, wave = tid >> 5;
    int mt = wave & 1;                                   // co tile (0/1)
    int w0 = (wave >> 1) << 4;                           // w tile base
    int lm = lane & 15, hi = lane >> 4;

    v8f acc = {};
    for (int t = 0; t < 27; ++t) {
        int kw = t % 3, kk = t / 3;                      // kk = kd*3+kh
        H16 A, B;
        const v8h* ap = (const v8h*)(wp + ((size_t)(t*2 + mt)*32 + lane) * 16);
        A.h[0] = ap[0]; A.h[1] = ap[1];
        const v8h* bp = (const v8h*)(tile + (kk*66 + (w0 + lm + kw)) * CG + hi * 8);
        B.h[0] = bp[0]; B.h[1] = bp[2];
        acc = __builtin_amdgcn_wmma_f32_16x16x32_f16(
            false, A.v, false, B.v, (short)0, acc, false, false);
    }
    // D layout (32-bit 16x16): lane<16 -> M=r, lane>=16 -> M=r+8; N = lane%16.
    float* zb = z + (size_t)g * CG * NSP + (size_t)d * (SZ*SZ) + h * SZ + (w0 + lm);
    #pragma unroll
    for (int r = 0; r < 8; ++r) {
        int co = mt * 16 + hi * 8 + r;
        zb[(size_t)co * NSP] = acc[r];
    }
}

// --- K5: per-(g,c) mean / rsqrt-var of conv output (b3 cancels, omitted).
__global__ __launch_bounds__(256) void k5_stats_z(const float* __restrict__ z,
                                                  float* __restrict__ m2,
                                                  float* __restrict__ i2) {
    int gc = blockIdx.x, tid = threadIdx.x;
    const float* base = z + (size_t)gc * NSP;
    float s1 = 0.f, s2 = 0.f;
    for (int n = tid; n < NSP; n += 256) { float v = base[n]; s1 += v; s2 += v*v; }
    __shared__ float r1[256], r2[256];
    r1[tid] = s1; r2[tid] = s2; __syncthreads();
    for (int off = 128; off > 0; off >>= 1) {
        if (tid < off) { r1[tid] += r1[tid+off]; r2[tid] += r2[tid+off]; }
        __syncthreads();
    }
    if (tid == 0) {
        float m = r1[0] / (float)NSP;
        m2[gc] = m; i2[gc] = rsqrtf(r2[0] / (float)NSP - m*m + EPSV);
    }
}

// --- K6: fused gate. r(n) = sum_c [ (gx*att - m1)*i1 + lrelu((z - m2)*i2) ];
//     out = gx * sigmoid(r). gx is L2-resident, so the re-read is cheap.
__global__ __launch_bounds__(64) void k6_final(const float* __restrict__ gx,
                                               const float* __restrict__ z,
                                               const float* __restrict__ sig,
                                               const float* __restrict__ m1,
                                               const float* __restrict__ i1,
                                               const float* __restrict__ m2,
                                               const float* __restrict__ i2,
                                               float* __restrict__ out) {
    int bid = blockIdx.x;
    int h = bid & 63, d = (bid >> 6) & 63, g = bid >> 12;
    int w = threadIdx.x;
    size_t nidx = (size_t)d * (SZ*SZ) + h * SZ + w;
    float gxv[CG];
    float r = 0.f;
    #pragma unroll
    for (int c = 0; c < CG; ++c) {
        int gc = g * CG + c;
        size_t off = (size_t)gc * NSP + nidx;
        float xv = gx[off];
        gxv[c] = xv;
        float att = sig[0*(NG*CG*SZ) + gc*SZ + d]
                  * sig[1*(NG*CG*SZ) + gc*SZ + h]
                  * sig[2*(NG*CG*SZ) + gc*SZ + w];
        float x1 = (xv * att - m1[gc]) * i1[gc];
        float zn = (z[off] - m2[gc]) * i2[gc];
        zn = (zn >= 0.f) ? zn : 0.01f * zn;
        r += x1 + zn;
    }
    float wgt = 1.f / (1.f + __expf(-r));
    #pragma unroll
    for (int c = 0; c < CG; ++c)
        out[(size_t)(g * CG + c) * NSP + nidx] = gxv[c] * wgt;
}

extern "C" void kernel_launch(void* const* d_in, const int* in_sizes, int n_in,
                              void* d_out, int out_size, void* d_ws, size_t ws_size,
                              hipStream_t stream) {
    const float* x  = (const float*)d_in[0];
    const float* w1 = (const float*)d_in[1];
    const float* b1 = (const float*)d_in[2];
    const float* w3 = (const float*)d_in[3];
    // b3 (d_in[4]) cancels inside instance_norm -> unused.

    float* wsf   = (float*)d_ws;
    float* z     = wsf + OFF_Z;
    float* pool  = wsf + OFF_POOL;
    float* sig   = wsf + OFF_SIG;
    float* m1    = wsf + OFF_M1;
    float* i1    = wsf + OFF_I1;
    float* m2    = wsf + OFF_M2;
    float* i2    = wsf + OFF_I2;
    _Float16* wp = (_Float16*)(wsf + OFF_WPK);
    float* out   = (float*)d_out;

    k0_repack_w3<<<(27*2*32*16 + 255) / 256, 256, 0, stream>>>(w3, wp);
    k1_pool<<<dim3(SZ, 3, NG*CG), 256, 0, stream>>>(x, pool);
    k2_mlp<<<NG, 256, 0, stream>>>(w1, b1, pool, sig);
    k3_stats_att<<<NG*CG, 256, 0, stream>>>(x, sig, m1, i1);
    k4_conv_wmma<<<NG*SZ*SZ, 256, 0, stream>>>(x, wp, z);
    k5_stats_z<<<NG*CG, 256, 0, stream>>>(z, m2, i2);
    k6_final<<<NG*SZ*SZ, 64, 0, stream>>>(x, z, sig, m1, i1, m2, i2, out);
}
